// NodeCoordEncoder_26594437496848
// MI455X (gfx1250) — compile-verified, matrix operand order checked by reading, not run on hardware
//
#include <hip/hip_runtime.h>
#include <hip/hip_bf16.h>
#include <math.h>

// ---------------------------------------------------------------------------
// GNN forward (TransformerConv x4 + BN + GCN heads) for MI455X / gfx1250.
// Dense K=64 GEMMs -> v_wmma_f32_16x16x32_bf16 with bf16 activations
// (pre-converted by the producing kernels) and pre-swizzled bf16 weight
// fragments: GEMM inner loop is pure b128 loads + WMMA.
// Edge phase -> L2-resident gathers + float atomics (segment softmax).
// ---------------------------------------------------------------------------

#define NN 50000
#define EE 500000
#define HC 256   // H*C
#define CC 64    // C
#define DL 32    // D_LAT

typedef __attribute__((ext_vector_type(16))) __bf16 v16bf;
typedef __attribute__((ext_vector_type(8)))  __bf16 v8bf;
typedef __attribute__((ext_vector_type(8)))  float  v8f;

__device__ __forceinline__ __bf16 f2bf(float f) {
  union { float f; unsigned u; } x; x.f = f;
  unsigned r = x.u + 0x7FFFu + ((x.u >> 16) & 1u);   // round-to-nearest-even
  union { unsigned short s; __bf16 b; } y; y.s = (unsigned short)(r >> 16);
  return y.b;
}

// ---------------- elementwise utilities ----------------
__global__ void fill_f32(float* __restrict__ p, float v, long n) {
  long i = (long)blockIdx.x * blockDim.x + threadIdx.x;
  if (i < n) p[i] = v;
}

// Pack W[K,Nc] (f32, row-major) into per-lane WMMA B fragments:
//   Bp[((nt*KT + kt)*32 + lane)*16 + j]  (bf16)
__global__ void pack_b(const float* __restrict__ W, __bf16* __restrict__ Bp,
                       int K, int Nc) {
  int idx = blockIdx.x * blockDim.x + threadIdx.x;
  int KT = K / 32;
  int total = (Nc / 16) * KT * 32;
  if (idx >= total) return;
  int lane = idx & 31;
  int frag = idx >> 5;              // nt*KT + kt
  int kt = frag % KT, nt = frag / KT;
  int half = lane >> 4;
  int n = nt * 16 + (lane & 15);
  __bf16* out = Bp + (size_t)idx * 16;
#pragma unroll
  for (int j = 0; j < 8; ++j) {
    int k1 = kt * 32 + half * 8 + j;
    int k2 = kt * 32 + 16 + half * 8 + j;
    out[j]     = f2bf(W[(long)k1 * Nc + n]);
    out[8 + j] = f2bf(W[(long)k2 * Nc + n]);
  }
}

// convert f32 matrix to bf16 (for layer-1 GEMM input path not needed; h only)
__global__ void cvt_f32_bf16(const float* __restrict__ s, __bf16* __restrict__ d, long n) {
  long i = (long)blockIdx.x * blockDim.x + threadIdx.x;
  if (i < n) d[i] = f2bf(s[i]);
}

// ---------------- WMMA GEMM: C[M,NC] = A[M,K](bf16) * Bpacked + bias --------
// One wave -> 16 rows x (TN*16) cols. K, NC compile-time; K % 32 == 0.
template <int K, int NC>
__global__ void gemm_wmma(const __bf16* __restrict__ A, const __bf16* __restrict__ Bp,
                          const float* __restrict__ bias, float* __restrict__ Cm,
                          int M) {
  constexpr int KT = K / 32;
  constexpr int NT_ALL = NC / 16;
  constexpr int TN = (NT_ALL < 4) ? NT_ALL : 4;
  int wave = threadIdx.x >> 5;
  int lane = threadIdx.x & 31;
  int row0 = (blockIdx.x * 8 + wave) * 16;
  if (row0 >= M) return;                       // wave-uniform exit
  int nt0  = blockIdx.y * TN;
  int half = lane >> 4;
  int m = row0 + (lane & 15);
  if (m >= M) m = M - 1;                       // clamp tail rows; stores guarded
  const __bf16* Arow  = A + (long)m * K;
  const v16bf*  Bfrag = (const v16bf*)Bp;

  v8f acc[TN];
#pragma unroll
  for (int t = 0; t < TN; ++t) acc[t] = v8f{};

#pragma unroll
  for (int kt = 0; kt < KT; ++kt) {
    const int k0 = kt * 32;
    v8bf lo = *(const v8bf*)(Arow + k0 + half * 8);        // 16B aligned
    v8bf hi = *(const v8bf*)(Arow + k0 + 16 + half * 8);   // 16B aligned
    v16bf a = __builtin_shufflevector(lo, hi, 0, 1, 2, 3, 4, 5, 6, 7,
                                      8, 9, 10, 11, 12, 13, 14, 15);
#pragma unroll
    for (int t = 0; t < TN; ++t) {
      v16bf b = Bfrag[((nt0 + t) * KT + kt) * 32 + lane];
      acc[t] = __builtin_amdgcn_wmma_f32_16x16x32_bf16(
          false, a, false, b, (short)0, acc[t], false, false);
    }
  }
#pragma unroll
  for (int t = 0; t < TN; ++t) {
    int n = (nt0 + t) * 16 + (lane & 15);
    float bb = bias ? bias[n] : 0.f;
#pragma unroll
    for (int r = 0; r < 8; ++r) {
      int mm = row0 + r + 8 * half;
      if (mm < M) Cm[(long)mm * NC + n] = acc[t][r] + bb;
    }
  }
}

// ---------------- naive GEMM for K=3 (layer 1) ----------------
__global__ void gemm_naive(const float* __restrict__ A, const float* __restrict__ W,
                           const float* __restrict__ bias, float* __restrict__ Cm,
                           int M, int K, int Nc) {
  long i = (long)blockIdx.x * blockDim.x + threadIdx.x;
  if (i >= (long)M * Nc) return;
  int mrow = (int)(i / Nc), n = (int)(i % Nc);
  float acc = bias ? bias[n] : 0.f;
  for (int k = 0; k < K; ++k) acc = fmaf(A[(long)mrow * K + k], W[(long)k * Nc + n], acc);
  Cm[i] = acc;
}

// ---------------- edge attention ----------------
__device__ __forceinline__ void atomicMaxF(float* addr, float val) {
  if (val >= 0.f) atomicMax((int*)addr, __float_as_int(val));
  else            atomicMin((unsigned int*)addr, __float_as_uint(val));
}

__global__ void edge_alpha(const float* __restrict__ q, const float* __restrict__ k,
                           const int* __restrict__ src, const int* __restrict__ dst,
                           float* __restrict__ alpha, float* __restrict__ mbuf, int E_) {
  long idx = (long)blockIdx.x * blockDim.x + threadIdx.x;
  if (idx >= (long)E_ * 4) return;
  int e = (int)(idx >> 2), h = (int)(idx & 3);
  int s = src[e], d = dst[e];
  const float4* qp = (const float4*)(q + (long)d * HC + h * CC);
  const float4* kp = (const float4*)(k + (long)s * HC + h * CC);
  float acc = 0.f;
#pragma unroll
  for (int i = 0; i < 16; ++i) {
    float4 a = qp[i], b = kp[i];
    acc = fmaf(a.x, b.x, acc); acc = fmaf(a.y, b.y, acc);
    acc = fmaf(a.z, b.z, acc); acc = fmaf(a.w, b.w, acc);
  }
  acc *= 0.125f;                         // 1/sqrt(C)
  alpha[idx] = acc;
  atomicMaxF(&mbuf[(long)d * 4 + h], acc);
}

__global__ void edge_exp(const float* __restrict__ mbuf, const int* __restrict__ dst,
                         float* __restrict__ alpha, float* __restrict__ denom, int E_) {
  long idx = (long)blockIdx.x * blockDim.x + threadIdx.x;
  if (idx >= (long)E_ * 4) return;
  int e = (int)(idx >> 2), h = (int)(idx & 3);
  int d = dst[e];
  float a = __expf(alpha[idx] - mbuf[(long)d * 4 + h]);
  alpha[idx] = a;
  atomicAdd(&denom[(long)d * 4 + h], a);
}

__global__ void edge_scatter(const float* __restrict__ alpha, const float* __restrict__ denom,
                             const float* __restrict__ v, const int* __restrict__ src,
                             const int* __restrict__ dst, float* __restrict__ agg, int E_) {
  long idx = (long)blockIdx.x * blockDim.x + threadIdx.x;
  if (idx >= (long)E_ * CC) return;
  int e = (int)(idx >> 6), c = (int)(idx & 63);
  int s = src[e], d = dst[e];
#pragma unroll
  for (int h = 0; h < 4; ++h) {
    float w = alpha[(long)e * 4 + h] / (denom[(long)d * 4 + h] + 1e-16f);
    atomicAdd(&agg[(long)d * HC + h * CC + c], w * v[(long)s * HC + h * CC + c]);
  }
}

// head-mean + beta gating + ReLU; writes f32 h (for BN stats) + bf16 h copy
__global__ void tconv_epilogue(const float* __restrict__ agg, const float* __restrict__ xr,
                               const float* __restrict__ Wb, float* __restrict__ hout,
                               __bf16* __restrict__ houtb, int M) {
  int n = blockIdx.x * blockDim.x + threadIdx.x;
  if (n >= M) return;
  const float* ap = agg + (long)n * HC;
  const float* xp = xr + (long)n * CC;
  float g_acc = 0.f;
#pragma unroll 4
  for (int c = 0; c < CC; ++c) {
    float o  = 0.25f * (ap[c] + ap[CC + c] + ap[2 * CC + c] + ap[3 * CC + c]);
    float xv = xp[c];
    g_acc = fmaf(o, Wb[c], g_acc);
    g_acc = fmaf(xv, Wb[CC + c], g_acc);
    g_acc = fmaf(o - xv, Wb[2 * CC + c], g_acc);
  }
  float g = 1.f / (1.f + __expf(-g_acc));
#pragma unroll 4
  for (int c = 0; c < CC; ++c) {
    float o = 0.25f * (ap[c] + ap[CC + c] + ap[2 * CC + c] + ap[3 * CC + c]);
    float val = fmaxf(g * xp[c] + (1.f - g) * o, 0.f);
    hout[(long)n * CC + c]  = val;
    houtb[(long)n * CC + c] = f2bf(val);
  }
}

// ---------------- BatchNorm (two-pass, LDS partials) ----------------
__global__ void bn_stats(const float* __restrict__ x, float* __restrict__ sums, int M) {
  __shared__ float s[2 * CC];
  for (int i = threadIdx.x; i < 2 * CC; i += blockDim.x) s[i] = 0.f;
  __syncthreads();
  long total = (long)M * CC;
  for (long i = (long)blockIdx.x * blockDim.x + threadIdx.x; i < total;
       i += (long)gridDim.x * blockDim.x) {
    int c = (int)(i & (CC - 1));
    float v = x[i];
    atomicAdd(&s[c], v);
    atomicAdd(&s[CC + c], v * v);
  }
  __syncthreads();
  for (int i = threadIdx.x; i < 2 * CC; i += blockDim.x) atomicAdd(&sums[i], s[i]);
}

__global__ void bn_apply(float* __restrict__ x, __bf16* __restrict__ xb,
                         const float* __restrict__ sums,
                         const float* __restrict__ gamma, const float* __restrict__ beta,
                         int M) {
  long i = (long)blockIdx.x * blockDim.x + threadIdx.x;
  if (i >= (long)M * CC) return;
  int c = (int)(i & (CC - 1));
  float invM = 1.f / (float)M;
  float mu  = sums[c] * invM;
  float var = sums[CC + c] * invM - mu * mu;
  float val = (x[i] - mu) * rsqrtf(var + 1e-5f) * gamma[c] + beta[c];
  x[i]  = val;
  xb[i] = f2bf(val);
}

// ---------------- GCN ----------------
__global__ void deg_kernel(const int* __restrict__ dst, float* __restrict__ deg, int E_) {
  int e = blockIdx.x * blockDim.x + threadIdx.x;
  if (e < E_) atomicAdd(&deg[dst[e]], 1.f);
}

__global__ void gcn_init(const float* __restrict__ xw, const float* __restrict__ deg,
                         const float* __restrict__ bias, float* __restrict__ out, int M) {
  long i = (long)blockIdx.x * blockDim.x + threadIdx.x;
  if (i >= (long)M * DL) return;
  int n = (int)(i / DL), c = (int)(i % DL);
  out[i] = bias[c] + xw[i] / deg[n];       // self-loop: dinv[n]^2 = 1/deg[n]
}

__global__ void gcn_scatter(const float* __restrict__ xw, const float* __restrict__ deg,
                            const int* __restrict__ src, const int* __restrict__ dst,
                            float* __restrict__ out, int E_) {
  long i = (long)blockIdx.x * blockDim.x + threadIdx.x;
  if (i >= (long)E_ * DL) return;
  int e = (int)(i / DL), c = (int)(i % DL);
  int s = src[e], d = dst[e];
  float norm = rsqrtf(deg[s]) * rsqrtf(deg[d]);
  atomicAdd(&out[(long)d * DL + c], norm * xw[(long)s * DL + c]);
}

// ---------------------------------------------------------------------------
extern "C" void kernel_launch(void* const* d_in, const int* in_sizes, int n_in,
                              void* d_out, int out_size, void* d_ws, size_t ws_size,
                              hipStream_t stream) {
  const float* x    = (const float*)d_in[0];
  const int*   ei   = (const int*)d_in[1];
  const float* Wq1  = (const float*)d_in[2];  const float* bq1  = (const float*)d_in[3];
  const float* Wk1  = (const float*)d_in[4];  const float* bk1  = (const float*)d_in[5];
  const float* Wv1  = (const float*)d_in[6];  const float* bv1  = (const float*)d_in[7];
  const float* Ws1  = (const float*)d_in[8];  const float* bs1  = (const float*)d_in[9];
  const float* Wb1  = (const float*)d_in[10];
  const float* Wq_r = (const float*)d_in[11]; const float* bq_r = (const float*)d_in[12];
  const float* Wk_r = (const float*)d_in[13]; const float* bk_r = (const float*)d_in[14];
  const float* Wv_r = (const float*)d_in[15]; const float* bv_r = (const float*)d_in[16];
  const float* Ws_r = (const float*)d_in[17]; const float* bs_r = (const float*)d_in[18];
  const float* Wb_r = (const float*)d_in[19];
  const float* bng  = (const float*)d_in[20]; const float* bnb  = (const float*)d_in[21];
  const float* Wmu  = (const float*)d_in[22]; const float* bmu  = (const float*)d_in[23];
  const float* Wlv  = (const float*)d_in[24]; const float* blv  = (const float*)d_in[25];
  const int* src = ei;
  const int* dst = ei + EE;

  // bump allocator over workspace
  char* base = (char*)d_ws; size_t off = 0;
  auto alloc = [&](size_t bytes) -> char* {
    char* p = base + off; off += (bytes + 255) & ~(size_t)255; return p;
  };
  float*  q    = (float*)alloc((size_t)NN * HC * 4);
  float*  kbuf = (float*)alloc((size_t)NN * HC * 4);
  float*  v    = (float*)alloc((size_t)NN * HC * 4);
  float*  agg  = (float*)alloc((size_t)NN * HC * 4);
  float*  hA   = (float*)alloc((size_t)NN * CC * 4);
  float*  hB   = (float*)alloc((size_t)NN * CC * 4);
  float*  xr   = (float*)alloc((size_t)NN * CC * 4);
  float*  alp  = (float*)alloc((size_t)EE * 4 * 4);
  float*  mbuf = (float*)alloc((size_t)NN * 4 * 4);
  float*  den  = (float*)alloc((size_t)NN * 4 * 4);
  float*  bns  = (float*)alloc(2 * CC * 4);
  float*  deg  = (float*)alloc((size_t)NN * 4);
  __bf16* hAb  = (__bf16*)alloc((size_t)NN * CC * 2);       // bf16 activation copies
  __bf16* hBb  = (__bf16*)alloc((size_t)NN * CC * 2);
  __bf16* WqB  = (__bf16*)alloc((size_t)3 * CC * HC * 2);   // packed fragments
  __bf16* WkB  = (__bf16*)alloc((size_t)3 * CC * HC * 2);
  __bf16* WvB  = (__bf16*)alloc((size_t)3 * CC * HC * 2);
  __bf16* WsB  = (__bf16*)alloc((size_t)3 * CC * CC * 2);
  __bf16* WmuB = (__bf16*)alloc((size_t)CC * DL * 2);
  __bf16* WlvB = (__bf16*)alloc((size_t)CC * DL * 2);

  // pack weights into WMMA fragment layout (once)
  for (int i = 0; i < 3; ++i) {
    size_t w1 = (size_t)i * CC * HC, w2 = (size_t)i * CC * CC;
    pack_b<<<(CC * HC / 16 + 255) / 256, 256, 0, stream>>>(Wq_r + w1, WqB + w1, CC, HC);
    pack_b<<<(CC * HC / 16 + 255) / 256, 256, 0, stream>>>(Wk_r + w1, WkB + w1, CC, HC);
    pack_b<<<(CC * HC / 16 + 255) / 256, 256, 0, stream>>>(Wv_r + w1, WvB + w1, CC, HC);
    pack_b<<<(CC * CC / 16 + 255) / 256, 256, 0, stream>>>(Ws_r + w2, WsB + w2, CC, CC);
  }
  pack_b<<<(CC * DL / 16 + 255) / 256, 256, 0, stream>>>(Wmu, WmuB, CC, DL);
  pack_b<<<(CC * DL / 16 + 255) / 256, 256, 0, stream>>>(Wlv, WlvB, CC, DL);

  auto run_attention = [&](const float* Wb, float* hout, __bf16* houtb) {
    fill_f32<<<(NN * 4 + 255) / 256, 256, 0, stream>>>(mbuf, -INFINITY, (long)NN * 4);
    fill_f32<<<(NN * 4 + 255) / 256, 256, 0, stream>>>(den, 0.f, (long)NN * 4);
    fill_f32<<<((long)NN * HC + 255) / 256, 256, 0, stream>>>(agg, 0.f, (long)NN * HC);
    edge_alpha<<<((long)EE * 4 + 255) / 256, 256, 0, stream>>>(q, kbuf, src, dst, alp, mbuf, EE);
    edge_exp<<<((long)EE * 4 + 255) / 256, 256, 0, stream>>>(mbuf, dst, alp, den, EE);
    edge_scatter<<<((long)EE * CC + 255) / 256, 256, 0, stream>>>(alp, den, v, src, dst, agg, EE);
    tconv_epilogue<<<(NN + 255) / 256, 256, 0, stream>>>(agg, xr, Wb, hout, houtb, NN);
  };
  auto run_bn = [&](float* xh, __bf16* xhb, const float* gamma, const float* beta) {
    fill_f32<<<1, 128, 0, stream>>>(bns, 0.f, 2 * CC);
    bn_stats<<<512, 256, 0, stream>>>(xh, bns, NN);
    bn_apply<<<((long)NN * CC + 255) / 256, 256, 0, stream>>>(xh, xhb, bns, gamma, beta, NN);
  };

  // ----- layer 1 (K = D_IN = 3 -> naive GEMMs) -----
  gemm_naive<<<((long)NN * HC + 255) / 256, 256, 0, stream>>>(x, Wq1, bq1, q, NN, 3, HC);
  gemm_naive<<<((long)NN * HC + 255) / 256, 256, 0, stream>>>(x, Wk1, bk1, kbuf, NN, 3, HC);
  gemm_naive<<<((long)NN * HC + 255) / 256, 256, 0, stream>>>(x, Wv1, bv1, v, NN, 3, HC);
  gemm_naive<<<((long)NN * CC + 255) / 256, 256, 0, stream>>>(x, Ws1, bs1, xr, NN, 3, CC);
  run_attention(Wb1, hA, hAb);
  run_bn(hA, hAb, bng, bnb);

  // ----- layers 2..4 (K = 64 -> WMMA GEMMs on bf16 activations) -----
  float*  cur  = hA;   __bf16* curb = hAb;
  float*  nxt  = hB;   __bf16* nxtb = hBb;
  const int MB = (NN + 127) / 128;   // 8 waves/block, 16 rows/wave
  for (int i = 0; i < 3; ++i) {
    size_t w1 = (size_t)i * CC * HC, w2 = (size_t)i * CC * CC;
    gemm_wmma<CC, HC><<<dim3(MB, 4), 256, 0, stream>>>(curb, WqB + w1, bq_r + i * HC, q, NN);
    gemm_wmma<CC, HC><<<dim3(MB, 4), 256, 0, stream>>>(curb, WkB + w1, bk_r + i * HC, kbuf, NN);
    gemm_wmma<CC, HC><<<dim3(MB, 4), 256, 0, stream>>>(curb, WvB + w1, bv_r + i * HC, v, NN);
    gemm_wmma<CC, CC><<<dim3(MB, 1), 256, 0, stream>>>(curb, WsB + w2, bs_r + i * CC, xr, NN);
    run_attention(Wb_r + i * 3 * CC, nxt, nxtb);
    if (i < 2) run_bn(nxt, nxtb, bng + (i + 1) * CC, bnb + (i + 1) * CC);
    float* t = cur; cur = nxt; nxt = t;
    __bf16* tb = curb; curb = nxtb; nxtb = tb;
  }

  // ----- GCN heads -----
  float* xwm = q;     // q/k buffers are free now
  float* xwl = kbuf;
  fill_f32<<<(NN + 255) / 256, 256, 0, stream>>>(deg, 1.f, NN);
  deg_kernel<<<(EE + 255) / 256, 256, 0, stream>>>(dst, deg, EE);
  gemm_wmma<CC, DL><<<dim3(MB, 1), 256, 0, stream>>>(curb, WmuB, nullptr, xwm, NN);
  gemm_wmma<CC, DL><<<dim3(MB, 1), 256, 0, stream>>>(curb, WlvB, nullptr, xwl, NN);

  float* outmu = (float*)d_out;
  float* outlv = outmu + (size_t)NN * DL;
  gcn_init<<<((long)NN * DL + 255) / 256, 256, 0, stream>>>(xwm, deg, bmu, outmu, NN);
  gcn_init<<<((long)NN * DL + 255) / 256, 256, 0, stream>>>(xwl, deg, blv, outlv, NN);
  gcn_scatter<<<((long)EE * DL + 255) / 256, 256, 0, stream>>>(xwm, deg, src, dst, outmu, EE);
  gcn_scatter<<<((long)EE * DL + 255) / 256, 256, 0, stream>>>(xwl, deg, src, dst, outlv, EE);
}
